// NTSNet_6992206758577
// MI455X (gfx1250) — matrix operand order repaired and checked by reading, not compiled, not run on hardware
//
#include <hip/hip_runtime.h>
#include <hip/hip_bf16.h>

typedef __attribute__((ext_vector_type(16))) _Float16 v16h;
typedef __attribute__((ext_vector_type(8)))  _Float16 v8h;
typedef __attribute__((ext_vector_type(8)))  float    v8f;

union H16 { v16h v; v8h h[2]; };

// ---------------- WMMA fragment loads (CDNA5 wave32 layouts) ----------------
// A (MxK, row-major, row = lane&15): lanes<16 hold K {k0..k0+7, k0+16..k0+23},
// lanes>=16 hold K {k0+8..k0+15, k0+24..k0+31}. Two 16B loads.
__device__ __forceinline__ v16h load_a_frag(const _Float16* row, int k0, int lane) {
    const _Float16* p = row + k0 + ((lane < 16) ? 0 : 8);
    H16 a;
    a.h[0] = *(const v8h*)(p);
    a.h[1] = *(const v8h*)(p + 16);
    return a.v;
}
// B stored [N][K] row-major (col = lane&15): lanes<16 read K {k0..k0+7, k0+16..k0+23},
// lanes>=16 read K {k0+8..k0+15, k0+24..k0+31} (same as A). Two 16B loads.
__device__ __forceinline__ v16h load_b_frag(const _Float16* col, int k0, int lane) {
    const _Float16* p = col + k0 + ((lane < 16) ? 0 : 8);
    H16 b;
    b.h[0] = *(const v8h*)(p);
    b.h[1] = *(const v8h*)(p + 16);
    return b.v;
}

#define WMMA_F16(a, b, c) \
    __builtin_amdgcn_wmma_f32_16x16x32_f16(false, (a), false, (b), (short)0, (c), false, false)

// ---------------- Register-blocked patch-embed GEMM ----------------
// Each wave computes a 32x64 output tile (2 M-subtiles x 4 N-subtiles):
// 12 b128 fragment loads feed 8 WMMAs per K-step (vs 4:1 unblocked).
// C[M][N] = relu(A[M][K] * B[N][K]^T + bias), optional f16 mirror Ch.
__global__ void gemm_block(const _Float16* __restrict__ A, const _Float16* __restrict__ B,
                           const float* __restrict__ bias, float* __restrict__ C,
                           _Float16* __restrict__ Ch, int M, int N, int K) {
    int wave = (int)((blockIdx.x * blockDim.x + threadIdx.x) >> 5);
    int lane = threadIdx.x & 31;
    int nblk = N >> 6;                 // 64-wide N blocks
    int mblk = (M + 31) >> 5;          // 32-tall M blocks
    int mb = wave / nblk, nb = wave % nblk;
    if (mb >= mblk) return;

    int row0 = mb * 32 + (lane & 15);
    int row1 = row0 + 16;
    const _Float16* arow0 = A + (size_t)(row0 < M ? row0 : 0) * K;
    const _Float16* arow1 = A + (size_t)(row1 < M ? row1 : 0) * K;
    const _Float16* bcol0 = B + (size_t)(nb * 64 + 0  + (lane & 15)) * K;
    const _Float16* bcol1 = B + (size_t)(nb * 64 + 16 + (lane & 15)) * K;
    const _Float16* bcol2 = B + (size_t)(nb * 64 + 32 + (lane & 15)) * K;
    const _Float16* bcol3 = B + (size_t)(nb * 64 + 48 + (lane & 15)) * K;

    v8f acc00 = {}, acc01 = {}, acc02 = {}, acc03 = {};
    v8f acc10 = {}, acc11 = {}, acc12 = {}, acc13 = {};

    for (int k0 = 0; k0 < K; k0 += 32) {
        if (k0 + 32 < K) {  // stream-ahead hints -> global_prefetch_b8
            __builtin_prefetch(arow0 + k0 + 32, 0, 3);
            __builtin_prefetch(arow1 + k0 + 32, 0, 3);
        }
        v16h a0 = load_a_frag(arow0, k0, lane);
        v16h a1 = load_a_frag(arow1, k0, lane);
        v16h b0 = load_b_frag(bcol0, k0, lane);
        v16h b1 = load_b_frag(bcol1, k0, lane);
        v16h b2 = load_b_frag(bcol2, k0, lane);
        v16h b3 = load_b_frag(bcol3, k0, lane);
        acc00 = WMMA_F16(a0, b0, acc00);
        acc10 = WMMA_F16(a1, b0, acc10);
        acc01 = WMMA_F16(a0, b1, acc01);
        acc11 = WMMA_F16(a1, b1, acc11);
        acc02 = WMMA_F16(a0, b2, acc02);
        acc12 = WMMA_F16(a1, b2, acc12);
        acc03 = WMMA_F16(a0, b3, acc03);
        acc13 = WMMA_F16(a1, b3, acc13);
    }

    v8f* accs[2][4] = { {&acc00, &acc01, &acc02, &acc03},
                        {&acc10, &acc11, &acc12, &acc13} };
#pragma unroll
    for (int j = 0; j < 4; ++j) {
        int col = nb * 64 + j * 16 + (lane & 15);
        float bv = bias ? bias[col] : 0.f;
#pragma unroll
        for (int i = 0; i < 2; ++i) {
            v8f a = *accs[i][j];
#pragma unroll
            for (int r = 0; r < 8; ++r) {
                int row = mb * 32 + i * 16 + r + ((lane < 16) ? 0 : 8);
                if (row < M) {
                    float v = a[r] + bv;
                    v = v > 0.f ? v : 0.f;
                    C[(size_t)row * N + col] = v;
                    if (Ch) Ch[(size_t)row * N + col] = (_Float16)v;
                }
            }
        }
    }
}

// ---------------- 3x3 conv as 9-tap accumulated WMMA GEMM ----------------
// in: [Bn][H][W][C] f16 (channels-last), wt: [N][9][C] f16, out: [Bn][OH][OW][N]
__global__ void conv3x3_wmma(const _Float16* __restrict__ in, const _Float16* __restrict__ wt,
                             const float* __restrict__ bias, float* __restrict__ out,
                             _Float16* __restrict__ outh,
                             int Bn, int H, int W, int C, int OH, int OW, int N, int stride) {
    int wave = (int)((blockIdx.x * blockDim.x + threadIdx.x) >> 5);
    int lane = threadIdx.x & 31;
    int ntiles = N >> 4;
    int M = Bn * OH * OW;
    int mtiles = (M + 15) >> 4;
    int mt = wave / ntiles, nt = wave % ntiles;
    if (mt >= mtiles) return;
    int rowm = mt * 16 + (lane & 15);
    bool rowOK = rowm < M;
    int mm = rowOK ? rowm : 0;
    int bi = mm / (OH * OW);
    int p  = mm % (OH * OW);
    int oy = p / OW, ox = p % OW;
    const _Float16* bcol = wt + (size_t)(nt * 16 + (lane & 15)) * 9 * C;
    v8f acc = {};
    for (int tap = 0; tap < 9; ++tap) {
        int ky = tap / 3, kx = tap % 3;
        int iy = oy * stride + ky - 1;
        int ix = ox * stride + kx - 1;
        bool ok = rowOK && iy >= 0 && iy < H && ix >= 0 && ix < W;
        const _Float16* arow = in + (size_t)((bi * H + (ok ? iy : 0)) * W + (ok ? ix : 0)) * C;
        const _Float16* brow = bcol + tap * C;
        for (int kk = 0; kk < C; kk += 32) {
            v16h a = {};
            if (ok) a = load_a_frag(arow, kk, lane);
            v16h bb = load_b_frag(brow, kk, lane);
            acc = WMMA_F16(a, bb, acc);
        }
    }
    int col = nt * 16 + (lane & 15);
    float bv = bias[col];
    for (int r = 0; r < 8; ++r) {
        int row = mt * 16 + r + ((lane < 16) ? 0 : 8);
        if (row < M) {
            float v = acc[r] + bv;
            v = v > 0.f ? v : 0.f;
            out[(size_t)row * N + col] = v;
            outh[(size_t)row * N + col] = (_Float16)v;
        }
    }
}

// ---------------- small helper kernels ----------------
__global__ void f32_to_f16(const float* __restrict__ src, _Float16* __restrict__ dst, int n) {
    int i = blockIdx.x * 256 + threadIdx.x;
    if (i < n) dst[i] = (_Float16)src[i];
}

// w [N][C][3][3] f32 -> out [N][9][C] f16
__global__ void pack3x3(const float* __restrict__ w, _Float16* __restrict__ out, int N, int C) {
    int i = blockIdx.x * 256 + threadIdx.x;
    int total = N * 9 * C;
    if (i >= total) return;
    int n = i / (9 * C), r = i % (9 * C), tap = r / C, c = r % C;
    out[i] = (_Float16)w[((size_t)n * C + c) * 9 + tap];
}

// im2col of main image: A[m=b*196+oy*14+ox][k=c*1024+ky*32+kx]
__global__ void pack_main(const float* __restrict__ x, _Float16* __restrict__ A) {
    int i = blockIdx.x * 256 + threadIdx.x;
    if (i >= 1568 * 3072) return;
    int m = i / 3072, k = i % 3072;
    int b = m / 196, p = m % 196, oy = p / 14, ox = p % 14;
    int c = k >> 10, r = k & 1023, ky = r >> 5, kx = r & 31;
    A[i] = (_Float16)x[(((size_t)b * 3 + c) * 448 + oy * 32 + ky) * 448 + ox * 32 + kx];
}

// anchors (1614 x 4 int, padded coords) — faithful port of _generate_default_anchor_maps
__global__ void gen_anchors(int* __restrict__ anch) {
    int i = blockIdx.x * 256 + threadIdx.x;
    if (i >= 1614) return;
    const double S13 = 1.2599210498948732;   // 2^(1/3)
    const double S23 = 1.5874010519681994;   // 2^(2/3)
    int base, per, ow, stride; double size;
    if (i < 1176)      { base = 0;    per = 196; ow = 14; stride = 32;  size = 48.0;  }
    else if (i < 1470) { base = 1176; per = 49;  ow = 7;  stride = 64;  size = 96.0;  }
    else               { base = 1470; per = 16;  ow = 4;  stride = 128; size = 192.0; }
    int local = i - base;
    int combo = local / per;
    int pos = local % per, gy = pos / ow, gx = pos % ow;
    int si = combo / 3, ai = combo % 3;
    double sc;
    if (i < 1470) sc = (si == 0) ? S13 : S23;
    else          sc = (si == 0) ? 1.0 : ((si == 1) ? S13 : S23);
    double ar = (ai == 0) ? 0.667 : ((ai == 1) ? 1.0 : 1.5);
    float cy = (float)(stride * 0.5 + stride * (double)gy);
    float cx = (float)(stride * 0.5 + stride * (double)gx);
    float h = (float)(size * sc / sqrt(ar));
    float w = (float)(size * sc * sqrt(ar));
    anch[i * 4 + 0] = (int)(cy - h * 0.5f + 224.0f);
    anch[i * 4 + 1] = (int)(cx - w * 0.5f + 224.0f);
    anch[i * 4 + 2] = (int)(cy + h * 0.5f + 224.0f);
    anch[i * 4 + 3] = (int)(cx + w * 0.5f + 224.0f);
}

// tier scores: rpn[b][off + o*HW + p] = tb[o] + sum_c y[b*HW+p][c] * tw[o][c]
__global__ void tier_scores(const float* __restrict__ y, const float* __restrict__ tw,
                            const float* __restrict__ tb, float* __restrict__ rpn,
                            int HW, int C, int O, int outOff) {
    int idx = blockIdx.x * 256 + threadIdx.x;
    int total = 8 * O * HW;
    if (idx >= total) return;
    int b = idx / (O * HW), r = idx % (O * HW), o = r / HW, p = r % HW;
    const float* yr = y + (size_t)(b * HW + p) * C;
    const float* wr = tw + (size_t)o * C;
    float s = tb[o];
    for (int c = 0; c < C; ++c) s += yr[c] * wr[c];
    rpn[b * 1614 + outOff + o * HW + p] = s;
}

#define NANCH 1614
// greedy hard NMS, TOP_N=4, per-batch block
__global__ void nms_kernel(const float* __restrict__ rpn, const int* __restrict__ anch,
                           int* __restrict__ boxes) {
    __shared__ float s[NANCH];
    __shared__ float rv[256];
    __shared__ int   ri[256];
    __shared__ float pbox[4];
    __shared__ float parea;
    int b = blockIdx.x, tid = threadIdx.x;
    for (int i = tid; i < NANCH; i += 256) s[i] = rpn[b * NANCH + i];
    __syncthreads();
    for (int it = 0; it < 4; ++it) {
        float bv = -3.4e38f; int biid = NANCH;
        for (int i = tid; i < NANCH; i += 256) {
            float v = s[i];
            if (v > bv || (v == bv && i < biid)) { bv = v; biid = i; }
        }
        rv[tid] = bv; ri[tid] = biid;
        __syncthreads();
        for (int off = 128; off > 0; off >>= 1) {
            if (tid < off) {
                float v2 = rv[tid + off]; int i2 = ri[tid + off];
                if (v2 > rv[tid] || (v2 == rv[tid] && i2 < ri[tid])) { rv[tid] = v2; ri[tid] = i2; }
            }
            __syncthreads();
        }
        if (tid == 0) {
            int idx = ri[0];
            for (int j = 0; j < 4; ++j) {
                int c = anch[idx * 4 + j];
                boxes[(b * 4 + it) * 4 + j] = c;
                pbox[j] = (float)c;
            }
            parea = (pbox[2] - pbox[0]) * (pbox[3] - pbox[1]);
        }
        __syncthreads();
        float py0 = pbox[0], px0 = pbox[1], py1 = pbox[2], px1 = pbox[3], pa = parea;
        for (int i = tid; i < NANCH; i += 256) {
            float y0 = (float)anch[i * 4 + 0], x0 = (float)anch[i * 4 + 1];
            float y1 = (float)anch[i * 4 + 2], x1 = (float)anch[i * 4 + 3];
            float sy = fmaxf(y0, py0), sx = fmaxf(x0, px0);
            float ey = fminf(y1, py1), ex = fminf(x1, px1);
            float ly = ey - sy, lx = ex - sx;
            float inter = (ly < 0.f || lx < 0.f) ? 0.f : ly * lx;
            float area = (y1 - y0) * (x1 - x0);
            float iou = inter / (area + pa - inter);
            if (iou >= 0.25f) s[i] = -3.4e38f;
        }
        __syncthreads();
    }
}

// fused bilinear crop-resize (align_corners, padded 896 coords) + im2col of parts
__global__ void parts_pack(const float* __restrict__ x, const int* __restrict__ boxes,
                           _Float16* __restrict__ A) {
    int idx = blockIdx.x * 256 + threadIdx.x;
    if (idx >= 1568 * 3072) return;
    int m = idx / 3072, k = idx % 3072;
    int part = m / 49, p = m % 49, oy = p / 7, ox = p % 7;
    int c = k >> 10, r = k & 1023, ky = r >> 5, kx = r & 31;
    int py = oy * 32 + ky, px = ox * 32 + kx;     // pixel in 224x224 crop
    int b = part >> 2, t = part & 3;
    const int* bx = boxes + (b * 4 + t) * 4;
    float y0 = (float)bx[0], x0 = (float)bx[1], y1 = (float)bx[2], x1 = (float)bx[3];
    float ys = y0 + (py / 223.0f) * (y1 - y0 - 1.0f);
    float xs = x0 + (px / 223.0f) * (x1 - x0 - 1.0f);
    float yf = floorf(ys), xf = floorf(xs);
    float wy = ys - yf, wx = xs - xf;
    int yl = (int)yf, xl = (int)xf;
    int yh = min(yl + 1, 895), xh = min(xl + 1, 895);
    const float* img = x + ((size_t)b * 3 + c) * 448 * 448;
    auto g = [&](int yy, int xx) -> float {
        yy -= 224; xx -= 224;   // padded -> original coords
        if (yy < 0 || yy >= 448 || xx < 0 || xx >= 448) return 0.f;
        return img[yy * 448 + xx];
    };
    float top = g(yl, xl) * (1.f - wx) + g(yl, xh) * wx;
    float bot = g(yh, xl) * (1.f - wx) + g(yh, xh) * wx;
    A[idx] = (_Float16)(top * (1.f - wy) + bot * wy);
}

// mean-pool part/raw features into concat [8][10240]
__global__ void pool_concat(const float* __restrict__ featPart, const float* __restrict__ featMain,
                            float* __restrict__ cc) {
    int idx = blockIdx.x * 256 + threadIdx.x;
    if (idx >= 8 * 10240) return;
    int b = idx / 10240, col = idx % 10240;
    float s = 0.f;
    if (col < 8192) {
        int t = col >> 11, c = col & 2047;
        const float* base = featPart + ((size_t)(b * 4 + t) * 49) * 2048 + c;
        for (int p = 0; p < 49; ++p) s += base[(size_t)p * 2048];
        s *= (1.0f / 49.0f);
    } else {
        int c = col - 8192;
        const float* base = featMain + ((size_t)b * 196) * 2048 + c;
        for (int p = 0; p < 196; ++p) s += base[(size_t)p * 2048];
        s *= (1.0f / 196.0f);
    }
    cc[idx] = s;
}

__global__ void final_linear(const float* __restrict__ cc, const float* __restrict__ cw,
                             const float* __restrict__ cb, float* __restrict__ out) {
    int idx = blockIdx.x * 256 + threadIdx.x;
    if (idx >= 1600) return;
    int b = idx / 200, o = idx % 200;
    const float* a = cc + (size_t)b * 10240;
    const float* w = cw + (size_t)o * 10240;
    float s = cb[o];
    for (int j = 0; j < 10240; ++j) s += a[j] * w[j];
    out[idx] = s;
}

// ---------------- launcher ----------------
extern "C" void kernel_launch(void* const* d_in, const int* in_sizes, int n_in,
                              void* d_out, int out_size, void* d_ws, size_t ws_size,
                              hipStream_t stream) {
    const float* x   = (const float*)d_in[0];
    const float* bbw = (const float*)d_in[1];
    const float* bbb = (const float*)d_in[2];
    const float* w1  = (const float*)d_in[3];
    const float* b1  = (const float*)d_in[4];
    const float* t1w = (const float*)d_in[5];
    const float* t1b = (const float*)d_in[6];
    const float* w2  = (const float*)d_in[7];
    const float* b2  = (const float*)d_in[8];
    const float* t2w = (const float*)d_in[9];
    const float* t2b = (const float*)d_in[10];
    const float* w3  = (const float*)d_in[11];
    const float* b3  = (const float*)d_in[12];
    const float* t3w = (const float*)d_in[13];
    const float* t3b = (const float*)d_in[14];
    const float* cw  = (const float*)d_in[15];
    const float* cb  = (const float*)d_in[16];
    float* out = (float*)d_out;

    char* wsb = (char*)d_ws;
    size_t off = 0;
    auto alloc = [&](size_t bytes) -> void* {
        void* p = wsb + off;
        off = (off + bytes + 255) & ~(size_t)255;
        return p;
    };
    int*       anch     = (int*)alloc(1614 * 4 * sizeof(int));
    float*     rpn      = (float*)alloc(8 * 1614 * sizeof(float));
    int*       boxes    = (int*)alloc(8 * 4 * 4 * sizeof(int));
    _Float16*  bbwh     = (_Float16*)alloc((size_t)2048 * 3072 * 2);
    _Float16*  w1h      = (_Float16*)alloc((size_t)128 * 9 * 2048 * 2);
    _Float16*  w2h      = (_Float16*)alloc((size_t)128 * 9 * 128 * 2);
    _Float16*  w3h      = (_Float16*)alloc((size_t)128 * 9 * 128 * 2);
    _Float16*  Apack    = (_Float16*)alloc((size_t)1568 * 3072 * 2);
    float*     featMain = (float*)alloc((size_t)1568 * 2048 * 4);
    _Float16*  featMainH= (_Float16*)alloc((size_t)1568 * 2048 * 2);
    float*     y1       = (float*)alloc((size_t)1568 * 128 * 4);
    _Float16*  y1h      = (_Float16*)alloc((size_t)1568 * 128 * 2);
    float*     y2       = (float*)alloc((size_t)392 * 128 * 4);
    _Float16*  y2h      = (_Float16*)alloc((size_t)392 * 128 * 2);
    float*     y3       = (float*)alloc((size_t)128 * 128 * 4);
    _Float16*  y3h      = (_Float16*)alloc((size_t)128 * 128 * 2);
    float*     featPart = (float*)alloc((size_t)1568 * 2048 * 4);
    float*     ccat     = (float*)alloc((size_t)8 * 10240 * 4);

    // weight conversions / packs
    f32_to_f16<<<(6291456 + 255) / 256, 256, 0, stream>>>(bbw, bbwh, 6291456);
    pack3x3<<<(2359296 + 255) / 256, 256, 0, stream>>>(w1, w1h, 128, 2048);
    pack3x3<<<(147456 + 255) / 256, 256, 0, stream>>>(w2, w2h, 128, 128);
    pack3x3<<<(147456 + 255) / 256, 256, 0, stream>>>(w3, w3h, 128, 128);
    gen_anchors<<<7, 256, 0, stream>>>(anch);

    // backbone on main image: im2col + register-blocked WMMA GEMM (M=1568,K=3072,N=2048)
    // waves = (1568/32) * (2048/64) = 49*32 = 1568 -> 196 blocks of 8 waves
    pack_main<<<(1568 * 3072) / 256, 256, 0, stream>>>(x, Apack);
    gemm_block<<<196, 256, 0, stream>>>(Apack, bbwh, bbb, featMain, featMainH,
                                        1568, 2048, 3072);

    // navigator
    conv3x3_wmma<<<98, 256, 0, stream>>>(featMainH, w1h, b1, y1, y1h, 8, 14, 14, 2048, 14, 14, 128, 1);
    tier_scores<<<(9408 + 255) / 256, 256, 0, stream>>>(y1, t1w, t1b, rpn, 196, 128, 6, 0);
    conv3x3_wmma<<<25, 256, 0, stream>>>(y1h, w2h, b2, y2, y2h, 8, 14, 14, 128, 7, 7, 128, 2);
    tier_scores<<<(2352 + 255) / 256, 256, 0, stream>>>(y2, t2w, t2b, rpn, 49, 128, 6, 1176);
    conv3x3_wmma<<<8, 256, 0, stream>>>(y2h, w3h, b3, y3, y3h, 8, 7, 7, 128, 4, 4, 128, 2);
    tier_scores<<<(1152 + 255) / 256, 256, 0, stream>>>(y3, t3w, t3b, rpn, 16, 128, 9, 1470);

    // NMS -> boxes; fused crop+im2col; backbone on parts (same GEMM shape)
    nms_kernel<<<8, 256, 0, stream>>>(rpn, anch, boxes);
    parts_pack<<<(1568 * 3072) / 256, 256, 0, stream>>>(x, boxes, Apack);
    gemm_block<<<196, 256, 0, stream>>>(Apack, bbwh, bbb, featPart, ((_Float16*)nullptr),
                                        1568, 2048, 3072);

    // pool, concat, classify
    pool_concat<<<(81920) / 256, 256, 0, stream>>>(featPart, featMain, ccat);
    final_linear<<<(1600 + 255) / 256, 256, 0, stream>>>(ccat, cw, cb, out);
}
